// MultiheadSelfAttention_51221779972522
// MI455X (gfx1250) — matrix operand order repaired and checked by reading, not compiled, not run on hardware
//
#include <hip/hip_runtime.h>

// ---------------------------------------------------------------------------
// Multi-head causal self-attention forward, MI455X (gfx1250), wave32 + WMMA.
// Pipeline: [qkv_rope] -> [attn_fwd (flash-style)] -> [out_proj]
// All GEMMs run on v_wmma_f32_16x16x32_bf16 with f32 accumulation.
// ---------------------------------------------------------------------------

typedef __bf16 bf16_t;
typedef __attribute__((ext_vector_type(16))) __bf16 v16bf;
typedef __attribute__((ext_vector_type(8)))  __bf16 v8bf;
typedef __attribute__((ext_vector_type(8)))  float  v8f;

constexpr int B_ = 4, S_ = 2048, D_ = 1024, H_ = 16, DK_ = 64;
constexpr float ROPE_C = -0.2878231366f;  // -ln(10000)/(DK/2)
constexpr float SCALE_ = 0.125f;          // 1/sqrt(DK)

static __device__ __forceinline__ v8f wmma_bf16(v16bf a, v16bf b, v8f c) {
  return __builtin_amdgcn_wmma_f32_16x16x32_bf16(false, a, false, b,
                                                 (short)0, c, false, false);
}

static __device__ __forceinline__ v16bf join16(v8bf lo, v8bf hi) {
  v16bf r;
#pragma unroll
  for (int e = 0; e < 8; ++e) { r[e] = lo[e]; r[e + 8] = hi[e]; }
  return r;
}

static __device__ __forceinline__ v8f vzero8() {
  v8f z;
#pragma unroll
  for (int i = 0; i < 8; ++i) z[i] = 0.0f;
  return z;
}

// ---------------------------------------------------------------------------
// Kernel 1: Q/K/V = x @ W^T (bf16 WMMA), RoPE on Q,K, V stored transposed.
//   grid = (B*S/16, 2, 3)  block = 256 (8 waves)
//   wave w handles head h = blockIdx.y*8 + w  (per-wave tile: 16 rows x 64 cols)
// A-layout (16x32 bf16): lane l, m = l&15; elems e -> K = kb+(l>>4)*16+e.
// B-layout (32x16 bf16): lane l, n = l&15; elems e -> K = kb+(l>>4)*16+e.
// C-layout (16x16 f32):  lane l, n = l&15; vgpr r -> m = (l>>4)*8 + r.
// ---------------------------------------------------------------------------
__global__ __launch_bounds__(256) void qkv_rope_kernel(
    const float* __restrict__ x,
    const float* __restrict__ Wq, const float* __restrict__ Wk,
    const float* __restrict__ Wv,
    bf16_t* __restrict__ Qb, bf16_t* __restrict__ Kb, bf16_t* __restrict__ Vt) {
  const int lane = threadIdx.x & 31;
  const int wave = threadIdx.x >> 5;
  const int lhi = lane >> 4, llo = lane & 15;

  const int row0 = blockIdx.x * 16;           // flat row in [0, B*S)
  const int b = row0 / S_;
  const int s0 = row0 % S_;
  const int h = blockIdx.y * 8 + wave;        // head
  const int z = blockIdx.z;                   // 0=Q 1=K 2=V
  const float* W = (z == 0) ? Wq : ((z == 1) ? Wk : Wv);

  const float* xrow = x + (size_t)(row0 + llo) * D_;

  v8f acc[4];
#pragma unroll
  for (int t = 0; t < 4; ++t) acc[t] = vzero8();

  for (int ks = 0; ks < D_ / 32; ++ks) {
    const int kb = ks * 32;
    // A tile from x (f32 -> bf16)
    v16bf a;
    {
      const float* p0 = xrow + kb + lhi * 16;
      const float* p1 = xrow + kb + lhi * 16 + 8;
#pragma unroll
      for (int e = 0; e < 8; ++e) {
        a[e]     = (bf16_t)p0[e];
        a[e + 8] = (bf16_t)p1[e];
      }
    }
#pragma unroll
    for (int t = 0; t < 4; ++t) {
      const float* wr =
          W + (size_t)(h * DK_ + t * 16 + llo) * D_ + kb + lhi * 16;
      v16bf bm;
#pragma unroll
      for (int e = 0; e < 16; ++e) bm[e] = (bf16_t)wr[e];
      acc[t] = wmma_bf16(a, bm, acc[t]);
    }
  }

  // RoPE (Q,K only): pairs (d, d+1) sit in adjacent lanes of C layout.
  if (z < 2) {
#pragma unroll
    for (int t = 0; t < 4; ++t) {
      const int d = t * 16 + llo;             // feature within head
      const float invf = __expf((float)(d >> 1) * ROPE_C);
      const bool even = (d & 1) == 0;
#pragma unroll
      for (int r = 0; r < 8; ++r) {
        const int srow = s0 + lhi * 8 + r;
        const float ang = (float)srow * invf;
        const float sn = __sinf(ang), cs = __cosf(ang);
        const float v = acc[t][r];
        const float pv = __shfl_xor(v, 1, 32);
        acc[t][r] = even ? (v * cs - pv * sn) : (v * cs + pv * sn);
      }
    }
  }

  // Store bf16. Q,K: [b,h,s,d].  V: transposed [b,h,d,s].
#pragma unroll
  for (int t = 0; t < 4; ++t) {
#pragma unroll
    for (int r = 0; r < 8; ++r) {
      const int srow = s0 + lhi * 8 + r;
      const int d = t * 16 + llo;
      const bf16_t val = (bf16_t)acc[t][r];
      if (z == 0)
        Qb[((size_t)(b * H_ + h) * S_ + srow) * DK_ + d] = val;
      else if (z == 1)
        Kb[((size_t)(b * H_ + h) * S_ + srow) * DK_ + d] = val;
      else
        Vt[((size_t)(b * H_ + h) * DK_ + d) * S_ + srow] = val;
    }
  }
}

// ---------------------------------------------------------------------------
// Kernel 2: flash-style causal attention. grid = (S/64, B*H), block = 128.
// Each wave owns 16 q rows; streams kv in 32-wide chunks with online softmax.
// P transposed C->A layout via a private per-wave LDS buffer (same-wave DS
// ordering is in-order; no block barrier so per-wave trip counts are safe).
// ---------------------------------------------------------------------------
__global__ __launch_bounds__(128) void attn_fwd_kernel(
    const bf16_t* __restrict__ Qb, const bf16_t* __restrict__ Kb,
    const bf16_t* __restrict__ Vt, bf16_t* __restrict__ Ab) {
  __shared__ __align__(32) bf16_t pbuf[4][16 * 32];

  const int lane = threadIdx.x & 31;
  const int wave = threadIdx.x >> 5;
  const int lhi = lane >> 4, llo = lane & 15;

  const int bh = blockIdx.y;
  const int b = bh / H_, h = bh % H_;
  const int q0 = blockIdx.x * 64 + wave * 16;

  const bf16_t* Qbase = Qb + (size_t)bh * S_ * DK_;
  const bf16_t* Kbase = Kb + (size_t)bh * S_ * DK_;
  const bf16_t* Vbase = Vt + (size_t)bh * DK_ * S_;

  // Q tile in A layout, both K-steps (DK = 64 = 2 x 32), kept in registers.
  v16bf qa[2];
#pragma unroll
  for (int ks = 0; ks < 2; ++ks) {
    const bf16_t* qp = Qbase + (size_t)(q0 + llo) * DK_ + ks * 32 + lhi * 16;
    qa[ks] = join16(*(const v8bf*)qp, *(const v8bf*)(qp + 8));
  }

  v8f o[4];
#pragma unroll
  for (int t = 0; t < 4; ++t) o[t] = vzero8();
  float mrun[8], lrun[8];
#pragma unroll
  for (int r = 0; r < 8; ++r) { mrun[r] = -1e30f; lrun[r] = 0.0f; }

  const int qmax = q0 + 15;
  for (int kv0 = 0; kv0 <= qmax; kv0 += 32) {
    // ---- scores S = Q @ K^T for this 16q x 32kv chunk (4 WMMAs) ----
    v8f sc[2];
#pragma unroll
    for (int nt = 0; nt < 2; ++nt) {
      sc[nt] = vzero8();
#pragma unroll
      for (int ks = 0; ks < 2; ++ks) {
        const bf16_t* kp = Kbase + (size_t)(kv0 + nt * 16 + llo) * DK_ +
                           ks * 32 + lhi * 16;
        sc[nt] = wmma_bf16(qa[ks], *(const v16bf*)kp, sc[nt]);
      }
    }

    // ---- scale, causal mask, online softmax (shuffle row reductions) ----
    float cf[8];
#pragma unroll
    for (int r = 0; r < 8; ++r) {
      const int qrow = q0 + lhi * 8 + r;
      float s0v = sc[0][r] * SCALE_;
      float s1v = sc[1][r] * SCALE_;
      if (kv0 + llo > qrow) s0v = -1e30f;
      if (kv0 + 16 + llo > qrow) s1v = -1e30f;
      float mx = fmaxf(s0v, s1v);
      mx = fmaxf(mx, __shfl_xor(mx, 1, 32));
      mx = fmaxf(mx, __shfl_xor(mx, 2, 32));
      mx = fmaxf(mx, __shfl_xor(mx, 4, 32));
      mx = fmaxf(mx, __shfl_xor(mx, 8, 32));
      const float mn = fmaxf(mrun[r], mx);
      cf[r] = __expf(mrun[r] - mn);
      const float p0 = __expf(s0v - mn);
      const float p1 = __expf(s1v - mn);
      sc[0][r] = p0;
      sc[1][r] = p1;
      float rs = p0 + p1;
      rs += __shfl_xor(rs, 1, 32);
      rs += __shfl_xor(rs, 2, 32);
      rs += __shfl_xor(rs, 4, 32);
      rs += __shfl_xor(rs, 8, 32);
      lrun[r] = lrun[r] * cf[r] + rs;
      mrun[r] = mn;
    }
#pragma unroll
    for (int t = 0; t < 4; ++t)
#pragma unroll
      for (int r = 0; r < 8; ++r) o[t][r] *= cf[r];

    // ---- transpose P (C layout -> A layout) through per-wave LDS ----
#pragma unroll
    for (int nt = 0; nt < 2; ++nt)
#pragma unroll
      for (int r = 0; r < 8; ++r)
        pbuf[wave][(lhi * 8 + r) * 32 + nt * 16 + llo] = (bf16_t)sc[nt][r];
    v16bf pa;
    {
      const bf16_t* pp = &pbuf[wave][llo * 32 + lhi * 16];
      pa = join16(*(const v8bf*)pp, *(const v8bf*)(pp + 8));
    }

    // ---- O += P @ V (4 WMMAs; V^T gives contiguous B-operand loads) ----
#pragma unroll
    for (int t = 0; t < 4; ++t) {
      const bf16_t* vp =
          Vbase + (size_t)(t * 16 + llo) * S_ + kv0 + lhi * 16;
      o[t] = wmma_bf16(pa, *(const v16bf*)vp, o[t]);
    }
  }

  // ---- finalize (1/l) and store attn output bf16 in [b, s, h*DK+d] ----
  float inv[8];
#pragma unroll
  for (int r = 0; r < 8; ++r) inv[r] = 1.0f / lrun[r];
#pragma unroll
  for (int t = 0; t < 4; ++t)
#pragma unroll
    for (int r = 0; r < 8; ++r) {
      const int srow = q0 + lhi * 8 + r;
      Ab[(size_t)(b * S_ + srow) * D_ + h * DK_ + t * 16 + llo] =
          (bf16_t)(o[t][r] * inv[r]);
    }
}

// ---------------------------------------------------------------------------
// Kernel 3: out = attn @ Wo^T (f32 output). grid = (B*S/16, 2), block = 256.
// ---------------------------------------------------------------------------
__global__ __launch_bounds__(256) void out_proj_kernel(
    const bf16_t* __restrict__ Ab, const float* __restrict__ Wo,
    float* __restrict__ out) {
  const int lane = threadIdx.x & 31;
  const int wave = threadIdx.x >> 5;
  const int lhi = lane >> 4, llo = lane & 15;

  const int row0 = blockIdx.x * 16;
  const int n0 = blockIdx.y * 512 + wave * 64;

  v8f acc[4];
#pragma unroll
  for (int t = 0; t < 4; ++t) acc[t] = vzero8();

  const bf16_t* arow = Ab + (size_t)(row0 + llo) * D_;
  for (int ks = 0; ks < D_ / 32; ++ks) {
    const int kb = ks * 32;
    const bf16_t* ap = arow + kb + lhi * 16;
    const v16bf a = join16(*(const v8bf*)ap, *(const v8bf*)(ap + 8));
#pragma unroll
    for (int t = 0; t < 4; ++t) {
      const float* wr =
          Wo + (size_t)(n0 + t * 16 + llo) * D_ + kb + lhi * 16;
      v16bf bm;
#pragma unroll
      for (int e = 0; e < 16; ++e) bm[e] = (bf16_t)wr[e];
      acc[t] = wmma_bf16(a, bm, acc[t]);
    }
  }

#pragma unroll
  for (int t = 0; t < 4; ++t)
#pragma unroll
    for (int r = 0; r < 8; ++r)
      out[(size_t)(row0 + lhi * 8 + r) * D_ + n0 + t * 16 + llo] = acc[t][r];
}

// ---------------------------------------------------------------------------
extern "C" void kernel_launch(void* const* d_in, const int* in_sizes, int n_in,
                              void* d_out, int out_size, void* d_ws,
                              size_t ws_size, hipStream_t stream) {
  const float* x  = (const float*)d_in[0];
  const float* Wq = (const float*)d_in[1];
  const float* Wk = (const float*)d_in[2];
  const float* Wv = (const float*)d_in[3];
  const float* Wo = (const float*)d_in[4];
  float* out = (float*)d_out;

  const size_t nBHSD = (size_t)B_ * H_ * S_ * DK_;  // 8 Mi elems
  bf16_t* Qb = (bf16_t*)d_ws;
  bf16_t* Kb = Qb + nBHSD;
  bf16_t* Vt = Kb + nBHSD;
  bf16_t* Ab = Vt + nBHSD;

  dim3 g1(B_ * S_ / 16, 2, 3);
  qkv_rope_kernel<<<g1, 256, 0, stream>>>(x, Wq, Wk, Wv, Qb, Kb, Vt);

  dim3 g2(S_ / 64, B_ * H_);
  attn_fwd_kernel<<<g2, 128, 0, stream>>>(Qb, Kb, Vt, Ab);

  dim3 g3(B_ * S_ / 16, 2);
  out_proj_kernel<<<g3, 256, 0, stream>>>(Ab, Wo, out);
}
